// DescrptDPA3S_21672404976278
// MI455X (gfx1250) — matrix operand order repaired and checked by reading, not compiled, not run on hardware
//
#include <hip/hip_runtime.h>
#include <math.h>

#define NLOC 1024
#define NALL 1024
#define E_SEL 120
#define A_SEL 40
#define N_DIM 128
#define E_DIM 64
#define A_DIM 32
#define AXIS 4
#define NLAYERS 6

#define D_NODE_IN 448   // 128 + 64*4 + 64
#define D_EDGE_IN 256   // 64 + 128 + 64

typedef __attribute__((ext_vector_type(16))) __bf16 v16bf;
typedef __attribute__((ext_vector_type(8)))  __bf16 v8bf;
typedef __attribute__((ext_vector_type(8)))  float  v8f;

__device__ __forceinline__ float siluf(float x) {
  return x * __builtin_amdgcn_rcpf(1.0f + __expf(-x));
}

__device__ __forceinline__ unsigned short f2bf(float f) {
  union { float f; unsigned u; } c; c.f = f;
  unsigned u = c.u;
  unsigned r = u + 0x7FFFu + ((u >> 16) & 1u);   // round-to-nearest-even
  return (unsigned short)(r >> 16);
}
__device__ __forceinline__ float switch_fn(float r, float rs, float rc) {
  float u = (r - rs) / (rc - rs);
  u = fminf(fmaxf(u, 0.0f), 1.0f);
  return ((-6.0f * u + 15.0f) * u - 10.0f) * u * u * u + 1.0f;
}
__device__ __forceinline__ v8f wmma_bf16(v16bf a, v16bf b, v8f c) {
  return __builtin_amdgcn_wmma_f32_16x16x32_bf16(false, a, false, b, (short)0, c,
                                                 false, false);
}
// A fragment from LDS row: two contiguous 8-elem (16B) chunks at +0 and +16
__device__ __forceinline__ v16bf frag_lds(const unsigned short* p) {
  v8bf a0 = *(const v8bf*)p;
  v8bf a1 = *(const v8bf*)(p + 16);
  return __builtin_shufflevector(a0, a1, 0, 1, 2, 3, 4, 5, 6, 7,
                                 8, 9, 10, 11, 12, 13, 14, 15);
}
// B fragment from pre-packed weights: one contiguous 16-elem (32B) run
__device__ __forceinline__ v16bf frag_pk(const unsigned short* p) {
  v8bf b0 = *(const v8bf*)p;
  v8bf b1 = *(const v8bf*)(p + 8);
  return __builtin_shufflevector(b0, b1, 0, 1, 2, 3, 4, 5, 6, 7,
                                 8, 9, 10, 11, 12, 13, 14, 15);
}

// ---------------- workspace layout (float element offsets) ----------------
#define OFF_EEBD   0ull                         // 1024*120*64
#define OFF_S      15728640ull                  // 1024*120
#define OFF_SWE    15851520ull                  // 1024*120
#define OFF_H      15974400ull                  // 1024*120*4
#define OFF_UVA    16465920ull                  // 1024*40*3
#define OFF_SWA    16588800ull                  // 1024*40
#define OFF_SWANG  16629760ull                  // 1024*40*40
#define OFF_AEBD   18268160ull                  // 1024*40*40*32
#define OFF_NODE   70696960ull                  // 1024*128
#define OFF_AMSG   70828032ull                  // 1024*40*32
#define OFF_EJ     72138752ull                  // 1024*40*32
#define FLOAT_TOTAL 73449472ull

#define NWPK_NODE  (NLAYERS * D_NODE_IN * N_DIM)   // 344064
#define NWPK_EDGE  (NLAYERS * D_EDGE_IN * E_DIM)   // 98304
#define NWPK_ANGLE (NLAYERS * A_DIM * A_DIM)       // 6144

// ---------------- one-time kernels ----------------

// Pack W[layer][K][N] (f32) into per-wave B-fragment order (bf16):
// out[layer][tile][ks][lane][e], K = ks*32 + (e&7) + 16*(e>>3) + 8*(lane>>4), n = tile*16 + (lane&15)
__global__ void k_wpack(const float* __restrict__ W, unsigned short* __restrict__ out,
                        int K, int N, int total) {
  int idx = blockIdx.x * blockDim.x + threadIdx.x;
  if (idx >= total) return;
  int KN = K * N;
  int layer = idx / KN;
  int r = idx % KN;
  int ksteps = K / 32;
  int tile = r / (ksteps * 512);
  int r2 = r % (ksteps * 512);
  int ks = r2 >> 9;
  int r3 = r2 & 511;
  int lane = r3 >> 4, e = r3 & 15;
  int half = lane >> 4, row = lane & 15;
  int kg = ks * 32 + (e & 7) + ((e >> 3) << 4) + (half << 3);
  int n = tile * 16 + row;
  out[idx] = f2bf(W[layer * KN + kg * N + n]);
}

__global__ void k_geom(const float* __restrict__ coord,
                       const int* __restrict__ nlist,
                       float* __restrict__ s, float* __restrict__ sw_e,
                       float* __restrict__ h, float* __restrict__ uva,
                       float* __restrict__ sw_a) {
  int idx = blockIdx.x * blockDim.x + threadIdx.x;
  if (idx >= NLOC * E_SEL) return;
  int i = idx / E_SEL, j = idx % E_SEL;
  int jn = nlist[idx];
  float dx = coord[jn * 3 + 0] - coord[i * 3 + 0];
  float dy = coord[jn * 3 + 1] - coord[i * 3 + 1];
  float dz = coord[jn * 3 + 2] - coord[i * 3 + 2];
  float r = sqrtf(dx * dx + dy * dy + dz * dz + 1e-12f);
  float swe = switch_fn(r, 5.0f, 6.0f);
  float inv = __builtin_amdgcn_rcpf(r);
  float ux = dx * inv, uy = dy * inv, uz = dz * inv;
  float sv = swe * __builtin_amdgcn_rcpf(r + 1e-6f);
  s[idx] = sv;
  sw_e[idx] = swe;
  h[idx * 4 + 0] = sv;
  h[idx * 4 + 1] = sv * ux;
  h[idx * 4 + 2] = sv * uy;
  h[idx * 4 + 3] = sv * uz;
  if (j < A_SEL) {
    uva[(i * A_SEL + j) * 3 + 0] = ux;
    uva[(i * A_SEL + j) * 3 + 1] = uy;
    uva[(i * A_SEL + j) * 3 + 2] = uz;
    sw_a[i * A_SEL + j] = switch_fn(r, 3.5f, 4.0f);
  }
}

__global__ void k_eebd_init(const float* __restrict__ s,
                            const float* __restrict__ ew,
                            const float* __restrict__ eb,
                            float* __restrict__ e_ebd) {
  int idx = blockIdx.x * blockDim.x + threadIdx.x;
  if (idx >= NLOC * E_SEL * E_DIM) return;
  int e = idx & (E_DIM - 1);
  int ij = idx / E_DIM;
  e_ebd[idx] = siluf(s[ij] * ew[e] + eb[e]);
}

__global__ void k_node_init(const int* __restrict__ atype,
                            const float* __restrict__ type_table,
                            float* __restrict__ node) {
  int idx = blockIdx.x * blockDim.x + threadIdx.x;
  if (idx >= NLOC * N_DIM) return;
  int i = idx / N_DIM, d = idx % N_DIM;
  node[idx] = type_table[atype[i] * N_DIM + d];
}

__global__ void k_angle_init(const float* __restrict__ uva,
                             const float* __restrict__ sw_a,
                             const float* __restrict__ aw,
                             const float* __restrict__ ab,
                             float* __restrict__ sw_ang,
                             float* __restrict__ a_ebd) {
  int idx = blockIdx.x * blockDim.x + threadIdx.x;   // (i, jj, kk)
  if (idx >= NLOC * A_SEL * A_SEL) return;
  int kk = idx % A_SEL;
  int jj = (idx / A_SEL) % A_SEL;
  int i = idx / (A_SEL * A_SEL);
  const float* uj = uva + (i * A_SEL + jj) * 3;
  const float* uk = uva + (i * A_SEL + kk) * 3;
  float cosang = uj[0] * uk[0] + uj[1] * uk[1] + uj[2] * uk[2];
  float sw = sw_a[i * A_SEL + jj] * sw_a[i * A_SEL + kk];
  sw_ang[idx] = sw;
  float* dst = a_ebd + (size_t)idx * A_DIM;
#pragma unroll
  for (int a = 0; a < A_DIM; ++a) dst[a] = siluf(cosang * aw[a] + ab[a]);
}

// ---------------- per-layer kernels ----------------

// node update: 16 atoms/block, 256 threads (8 waves), WMMA 16x448x128
__global__ void k_node_layer(const float* __restrict__ e_ebd,
                             const float* __restrict__ sw_e,
                             const float* __restrict__ h,
                             float* __restrict__ node,
                             const unsigned short* __restrict__ nodeW_pk,
                             const float* __restrict__ node_b, int layer) {
  __shared__ float sh_gr[16 * E_DIM * 4];
  __shared__ float sh_em[16 * E_DIM];
  __shared__ unsigned short sh_y[16 * D_NODE_IN];

  int tid = threadIdx.x;
  int aL = tid >> 4;
  int l16 = tid & 15;
  int e0 = l16 * 4;
  int atom = blockIdx.x * 16 + aL;

  float gr[4][4];
  float em[4];
#pragma unroll
  for (int de = 0; de < 4; ++de) {
    em[de] = 0.f;
#pragma unroll
    for (int x = 0; x < 4; ++x) gr[de][x] = 0.f;
  }
  const float* ebase = e_ebd + (size_t)atom * E_SEL * E_DIM;
  const float* swb = sw_e + atom * E_SEL;
  const float* hb = h + atom * E_SEL * 4;
  for (int j = 0; j < E_SEL; ++j) {
    float sw = swb[j];
    float h0 = hb[j * 4 + 0], h1 = hb[j * 4 + 1], h2 = hb[j * 4 + 2], h3 = hb[j * 4 + 3];
    const float* ep = ebase + j * E_DIM + e0;
#pragma unroll
    for (int de = 0; de < 4; ++de) {
      float ge = ep[de] * sw;
      em[de] += ge;
      gr[de][0] += ge * h0; gr[de][1] += ge * h1;
      gr[de][2] += ge * h2; gr[de][3] += ge * h3;
    }
  }
  const float inv_seln = 1.0f / 12.0f;      // SEL_NORM
  const float inv_sel = 1.0f / (float)E_SEL;
#pragma unroll
  for (int de = 0; de < 4; ++de) {
    sh_em[aL * E_DIM + e0 + de] = em[de] * inv_sel;
#pragma unroll
    for (int x = 0; x < 4; ++x)
      sh_gr[(aL * E_DIM + e0 + de) * 4 + x] = gr[de][x] * inv_seln;
  }
  __syncthreads();

#pragma unroll
  for (int q = 0; q < 8; ++q) {
    int d = l16 * 8 + q;
    sh_y[aL * D_NODE_IN + d] = f2bf(node[atom * N_DIM + d]);
  }
#pragma unroll
  for (int de = 0; de < 4; ++de) {
    int e = e0 + de;
    const float* ge_ = &sh_gr[(aL * E_DIM + e) * 4];
#pragma unroll
    for (int ax = 0; ax < AXIS; ++ax) {
      const float* ga_ = &sh_gr[(aL * E_DIM + ax) * 4];
      float v = ge_[0] * ga_[0] + ge_[1] * ga_[1] + ge_[2] * ga_[2] + ge_[3] * ga_[3];
      sh_y[aL * D_NODE_IN + N_DIM + e * 4 + ax] = f2bf(v);
    }
    sh_y[aL * D_NODE_IN + N_DIM + 256 + e] = f2bf(sh_em[aL * E_DIM + e]);
  }
  __syncthreads();

  int wave = tid >> 5, lane = tid & 31;
  int half = lane >> 4, row = lane & 15;
  int n0 = wave * 16;
  const unsigned short* Wp =
      nodeW_pk + (size_t)layer * D_NODE_IN * N_DIM + ((size_t)wave * 14) * 512 + (lane << 4);
  v8f c = {};
#pragma unroll 2
  for (int ks = 0; ks < D_NODE_IN / 32; ++ks) {
    v16bf a = frag_lds(&sh_y[row * D_NODE_IN + ks * 32 + (half << 3)]);
    v16bf b = frag_pk(Wp + (size_t)ks * 512);
    c = wmma_bf16(a, b, c);
  }
#pragma unroll
  for (int v = 0; v < 8; ++v) {
    int M = v + (half << 3);
    int N = n0 + row;
    int at = blockIdx.x * 16 + M;
    float old = node[at * N_DIM + N];
    node[at * N_DIM + N] = old + siluf(c[v] + node_b[layer * N_DIM + N]);
  }
}

// edge MLP + fused a_msg add + e_ebd accumulate.
// blocks = (atom, mtile 0..7); 128 threads (4 waves over N=64)
__global__ void k_edge_layer(float* __restrict__ e_ebd,
                             const float* __restrict__ node,
                             const int* __restrict__ nlist,
                             const float* __restrict__ amsg,
                             const float* __restrict__ ea_w,
                             const float* __restrict__ sw_e,
                             const unsigned short* __restrict__ edgeW_pk,
                             const float* __restrict__ edge_b, int layer) {
  __shared__ unsigned short shA[16 * D_EDGE_IN];
  __shared__ float shMsg[16 * A_DIM];
  int bx = blockIdx.x;
  int i = bx >> 3, mt = bx & 7;
  int tid = threadIdx.x;
  bool hasA = (mt * 16) < A_SEL;

  for (int idx = tid; idx < 16 * D_EDGE_IN; idx += 128) {
    int rowm = idx >> 8, col = idx & 255;
    int j = mt * 16 + rowm;
    unsigned short v = 0;
    if (j < E_SEL) {
      float f;
      if (col < E_DIM)              f = e_ebd[((size_t)i * E_SEL + j) * E_DIM + col];
      else if (col < E_DIM + N_DIM) f = node[i * N_DIM + (col - E_DIM)];
      else { int jn = nlist[i * E_SEL + j]; f = node[jn * N_DIM + (col - E_DIM - N_DIM)]; }
      v = f2bf(f);
    }
    shA[idx] = v;
  }
  if (hasA) {
    for (int idx = tid; idx < 16 * A_DIM; idx += 128) {
      int rowm = idx >> 5, b = idx & 31;
      int j = mt * 16 + rowm;
      shMsg[idx] = (j < A_SEL) ? amsg[(i * A_SEL + j) * A_DIM + b] : 0.f;
    }
  }
  __syncthreads();

  int wave = tid >> 5, lane = tid & 31;
  int half = lane >> 4, row = lane & 15;
  int n0 = wave * 16;
  const unsigned short* Wp =
      edgeW_pk + (size_t)layer * D_EDGE_IN * E_DIM + ((size_t)wave * 8) * 512 + (lane << 4);
  v8f c = {};
#pragma unroll
  for (int ks = 0; ks < D_EDGE_IN / 32; ++ks) {
    v16bf a = frag_lds(&shA[row * D_EDGE_IN + ks * 32 + (half << 3)]);
    v16bf b = frag_pk(Wp + (size_t)ks * 512);
    c = wmma_bf16(a, b, c);
  }

  float eac[A_DIM];
  if (hasA) {
    const float* W = ea_w + (size_t)layer * A_DIM * E_DIM;
#pragma unroll
    for (int b = 0; b < A_DIM; ++b) eac[b] = W[b * E_DIM + n0 + row];
  }
#pragma unroll
  for (int v = 0; v < 8; ++v) {
    int M = v + (half << 3);
    int N = n0 + row;
    int j = mt * 16 + M;
    if (j < E_SEL) {
      float upd = siluf(c[v] + edge_b[layer * E_DIM + N]);
      if (hasA && j < A_SEL) {
        float acc = 0.f;
#pragma unroll
        for (int b = 0; b < A_DIM; ++b) acc += shMsg[M * A_DIM + b] * eac[b];
        upd += acc;
      }
      size_t idx = ((size_t)i * E_SEL + j) * E_DIM + N;
      e_ebd[idx] += upd * sw_e[i * E_SEL + j];
    }
  }
}

// a_msg = mean_k(a_ebd * sw_ang)
__global__ void k_amsg(const float* __restrict__ a_ebd,
                       const float* __restrict__ sw_ang,
                       float* __restrict__ amsg) {
  int idx = blockIdx.x * blockDim.x + threadIdx.x;
  if (idx >= NLOC * A_SEL * A_DIM) return;
  int a = idx & (A_DIM - 1);
  int jj = (idx / A_DIM) % A_SEL;
  int i = idx / (A_DIM * A_SEL);
  const float* ap = a_ebd + ((size_t)(i * A_SEL + jj) * A_SEL) * A_DIM + a;
  const float* sp = sw_ang + (i * A_SEL + jj) * A_SEL;
  float acc = 0.f;
  for (int k = 0; k < A_SEL; ++k) acc += ap[(size_t)k * A_DIM] * sp[k];
  amsg[idx] = acc * (1.0f / (float)A_SEL);
}

// ej = e_ebd[:, :A_SEL] @ angle_ew
__global__ void k_ej(const float* __restrict__ e_ebd,
                     const float* __restrict__ angle_ew,
                     float* __restrict__ ej, int layer) {
  int idx = blockIdx.x * blockDim.x + threadIdx.x;
  if (idx >= NLOC * A_SEL * A_DIM) return;
  int a = idx & (A_DIM - 1);
  int jj = (idx / A_DIM) % A_SEL;
  int i = idx / (A_DIM * A_SEL);
  const float* ep = e_ebd + ((size_t)i * E_SEL + jj) * E_DIM;
  const float* W = angle_ew + (size_t)layer * E_DIM * A_DIM + a;
  float acc = 0.f;
#pragma unroll
  for (int e = 0; e < E_DIM; ++e) acc += ep[e] * W[e * A_DIM];
  ej[idx] = acc;
}

// angle update: blocks = (atom, ptile 0..99); 64 threads (2 waves over N=32)
__global__ void k_angle_layer(float* __restrict__ a_ebd,
                              const float* __restrict__ ej,
                              const float* __restrict__ sw_ang,
                              const unsigned short* __restrict__ angleW_pk,
                              const float* __restrict__ angle_b, int layer) {
  __shared__ unsigned short shA[16 * A_DIM];
  int bx = blockIdx.x;
  int i = bx / 100, pt = bx % 100;
  int tid = threadIdx.x;
  size_t rbase = ((size_t)i * (A_SEL * A_SEL) + pt * 16);

  for (int idx = tid; idx < 16 * A_DIM; idx += 64) {
    int rowm = idx >> 5, col = idx & 31;
    shA[idx] = f2bf(a_ebd[(rbase + rowm) * A_DIM + col]);
  }
  __syncthreads();

  int wave = tid >> 5, lane = tid & 31;
  int half = lane >> 4, row = lane & 15;
  int n0 = wave * 16;
  const unsigned short* Wp =
      angleW_pk + layer * A_DIM * A_DIM + (wave << 9) + (lane << 4);
  v16bf a = frag_lds(&shA[row * A_DIM + (half << 3)]);
  v16bf b = frag_pk(Wp);
  v8f c = {};
  c = wmma_bf16(a, b, c);

#pragma unroll
  for (int v = 0; v < 8; ++v) {
    int M = v + (half << 3);
    int N = n0 + row;
    int p = pt * 16 + M;
    int jj = p / A_SEL, kk = p % A_SEL;
    size_t idx = ((size_t)i * (A_SEL * A_SEL) + p) * A_DIM + N;
    float t = c[v] + ej[(i * A_SEL + jj) * A_DIM + N] + ej[(i * A_SEL + kk) * A_DIM + N] +
              angle_b[layer * A_DIM + N];
    a_ebd[idx] += siluf(t) * sw_ang[i * (A_SEL * A_SEL) + p];
  }
}

// ---------------- host launcher ----------------
extern "C" void kernel_launch(void* const* d_in, const int* in_sizes, int n_in,
                              void* d_out, int out_size, void* d_ws, size_t ws_size,
                              hipStream_t stream) {
  (void)in_sizes; (void)n_in; (void)out_size; (void)ws_size;
  const float* coord      = (const float*)d_in[0];
  const int*   atype      = (const int*)d_in[1];
  const int*   nlist      = (const int*)d_in[2];
  const float* type_table = (const float*)d_in[3];
  const float* edge_in_w  = (const float*)d_in[4];
  const float* edge_in_b  = (const float*)d_in[5];
  const float* angle_in_w = (const float*)d_in[6];
  const float* angle_in_b = (const float*)d_in[7];
  const float* node_w     = (const float*)d_in[8];
  const float* node_b     = (const float*)d_in[9];
  const float* edge_w     = (const float*)d_in[10];
  const float* edge_b     = (const float*)d_in[11];
  const float* angle_w    = (const float*)d_in[12];
  const float* angle_ew   = (const float*)d_in[13];
  const float* angle_b    = (const float*)d_in[14];
  const float* ea_w       = (const float*)d_in[15];

  float* F = (float*)d_ws;
  float* e_ebd  = F + OFF_EEBD;
  float* s      = F + OFF_S;
  float* sw_e   = F + OFF_SWE;
  float* h      = F + OFF_H;
  float* uva    = F + OFF_UVA;
  float* sw_a   = F + OFF_SWA;
  float* sw_ang = F + OFF_SWANG;
  float* a_ebd  = F + OFF_AEBD;
  float* node   = F + OFF_NODE;
  float* amsg   = F + OFF_AMSG;
  float* ej     = F + OFF_EJ;
  unsigned short* nodeW_pk  = (unsigned short*)((char*)d_ws + FLOAT_TOTAL * 4);
  unsigned short* edgeW_pk  = nodeW_pk + NWPK_NODE;
  unsigned short* angleW_pk = edgeW_pk + NWPK_EDGE;

  // one-time setup: pack weights into WMMA B-fragment order (bf16)
  k_wpack<<<(NWPK_NODE + 255) / 256, 256, 0, stream>>>(node_w, nodeW_pk, D_NODE_IN, N_DIM,
                                                       NWPK_NODE);
  k_wpack<<<(NWPK_EDGE + 255) / 256, 256, 0, stream>>>(edge_w, edgeW_pk, D_EDGE_IN, E_DIM,
                                                       NWPK_EDGE);
  k_wpack<<<(NWPK_ANGLE + 255) / 256, 256, 0, stream>>>(angle_w, angleW_pk, A_DIM, A_DIM,
                                                        NWPK_ANGLE);

  k_geom<<<(NLOC * E_SEL + 127) / 128, 128, 0, stream>>>(coord, nlist, s, sw_e, h, uva, sw_a);
  k_eebd_init<<<(NLOC * E_SEL * E_DIM + 255) / 256, 256, 0, stream>>>(s, edge_in_w,
                                                                      edge_in_b, e_ebd);
  k_node_init<<<(NLOC * N_DIM + 255) / 256, 256, 0, stream>>>(atype, type_table, node);
  k_angle_init<<<(NLOC * A_SEL * A_SEL + 255) / 256, 256, 0, stream>>>(
      uva, sw_a, angle_in_w, angle_in_b, sw_ang, a_ebd);

  for (int l = 0; l < NLAYERS; ++l) {
    k_node_layer<<<NLOC / 16, 256, 0, stream>>>(e_ebd, sw_e, h, node, nodeW_pk, node_b, l);
    k_amsg<<<(NLOC * A_SEL * A_DIM + 255) / 256, 256, 0, stream>>>(a_ebd, sw_ang, amsg);
    k_edge_layer<<<NLOC * 8, 128, 0, stream>>>(e_ebd, node, nlist, amsg, ea_w, sw_e,
                                               edgeW_pk, edge_b, l);
    k_ej<<<(NLOC * A_SEL * A_DIM + 255) / 256, 256, 0, stream>>>(e_ebd, angle_ew, ej, l);
    k_angle_layer<<<NLOC * 100, 64, 0, stream>>>(a_ebd, ej, sw_ang, angleW_pk, angle_b, l);
  }

  hipMemcpyAsync(d_out, node, (size_t)NLOC * N_DIM * sizeof(float),
                 hipMemcpyDeviceToDevice, stream);
}
